// AlignMutualInfo_18090402251042
// MI455X (gfx1250) — matrix operand order repaired and checked by reading, not compiled, not run on hardware
//
#include <hip/hip_runtime.h>
#include <stdint.h>

// ---------------------------------------------------------------------------
// AlignMutualInfo fused loss for gfx1250 (MI455X, wave32, WMMA).
//   L  = lm_embeds[ N x 1024] @ lm_W [1024x128] + lm_b
//   Gp = gnn_embeds[N x 200 ] @ gnn_W[ 200x128] + gnn_b   (Gn with neg embeds)
//   per-row: cos = (l.g)/(|l||g|); e=exp(cos); ratio=e/(e+C)
//   out = -mean(log ratio_pos) - mean(log(1-ratio_neg)),  C = 1/11008
// GEMMs on v_wmma_f32_16x16x32_bf16. 32 rows/block: each B fragment loaded
// from L2 once feeds two WMMAs (two row-tiles). A fragments staged in LDS.
// ---------------------------------------------------------------------------

typedef __attribute__((ext_vector_type(16))) __bf16 v16bf;
typedef __attribute__((ext_vector_type(2)))  __bf16 v2bf;
typedef __attribute__((ext_vector_type(8)))  float  v8f;

union BfPack {
    unsigned u32[8];
    uint4    u128[2];
    v16bf    v;
};

// Pack two f32 -> one dword of two bf16 (RNE). Prefer the HW packed convert.
__device__ __forceinline__ unsigned pack2bf(float x, float y) {
#if defined(__has_builtin)
#if __has_builtin(__builtin_amdgcn_cvt_pk_bf16_f32)
    union { v2bf v; unsigned u; } cv;
    cv.v = __builtin_amdgcn_cvt_pk_bf16_f32(x, y);
    return cv.u;
#else
    union { __bf16 h[2]; unsigned u; } cv;
    cv.h[0] = (__bf16)x; cv.h[1] = (__bf16)y;
    return cv.u;
#endif
#else
    union { __bf16 h[2]; unsigned u; } cv;
    cv.h[0] = (__bf16)x; cv.h[1] = (__bf16)y;
    return cv.u;
#endif
}

// ---------------------------------------------------------------------------
// Pack f32 weight matrix W[K x 128] into bf16 B-operand tiles (L2-resident).
// Tile (kt, nt): K rows kt*32..+31, cols nt*16..+15. V_WMMA B layout (32x16):
// lane l -> column (l&15), K base = (l>=16 ? 8 : 0), K = base..+7 and base+16..+23 packed
// 2-per-dword. One wave packs one tile.
// ---------------------------------------------------------------------------
__global__ void __launch_bounds__(256)
pack_weights_kernel(const float* __restrict__ W, int K, unsigned* __restrict__ out) {
    const int kt   = blockIdx.x;
    const int nt   = threadIdx.x >> 5;
    const int lane = threadIdx.x & 31;
    const int half = lane >> 4;
    const int l16  = lane & 15;
    const int col   = nt * 16 + l16;
    const int kbase = kt * 32 + half * 8;

    float t[16];
#pragma unroll
    for (int i = 0; i < 16; ++i) {
        const int k = kbase + (i < 8 ? i : i + 8);
        t[i] = (k < K) ? W[(size_t)k * 128 + col] : 0.0f;
    }
    BfPack p;
#pragma unroll
    for (int j = 0; j < 8; ++j) p.u32[j] = pack2bf(t[2 * j], t[2 * j + 1]);

    uint4* dst = reinterpret_cast<uint4*>(out) +
                 ((size_t)(kt * 8 + nt) * 32 + lane) * 2;
    dst[0] = p.u128[0];
    dst[1] = p.u128[1];
}

// ---------------------------------------------------------------------------
// Stage bf16 A fragments for one 16-row tile into LDS (cooperative, 256 thr).
// Packet p = (kt, lane): 16-bit A layout -> lane holds K = ks..ks+7 and
// ks+16..ks+23 with ks = kt*32 + (lane>=16 ? 8 : 0), row = rowbase + (lane&15).
// ---------------------------------------------------------------------------
__device__ __forceinline__ void stage_A(const float* __restrict__ M,
                                        int rowStride, int K, int nkt,
                                        int rowbase, uint4* __restrict__ lds) {
    const int npack = nkt * 32;
    for (int p = threadIdx.x; p < npack; p += 256) {
        const int kt   = p >> 5;
        const int lane = p & 31;
        const int half = lane >> 4;
        const int l16  = lane & 15;
        const float* rowp = M + (size_t)(rowbase + l16) * rowStride;
        const int ks = kt * 32 + half * 8;
        BfPack fr;
        if (ks + 23 < K) {
            const float4* p4 = reinterpret_cast<const float4*>(rowp + ks);
            const float4 x0 = p4[0], x1 = p4[1], y0 = p4[4], y1 = p4[5];
            fr.u32[0] = pack2bf(x0.x, x0.y); fr.u32[1] = pack2bf(x0.z, x0.w);
            fr.u32[2] = pack2bf(x1.x, x1.y); fr.u32[3] = pack2bf(x1.z, x1.w);
            fr.u32[4] = pack2bf(y0.x, y0.y); fr.u32[5] = pack2bf(y0.z, y0.w);
            fr.u32[6] = pack2bf(y1.x, y1.y); fr.u32[7] = pack2bf(y1.z, y1.w);
        } else {
            float t[16];
#pragma unroll
            for (int i = 0; i < 8; ++i) {
                const int k0 = ks + i, k1 = ks + 16 + i;
                t[i]     = (k0 < K) ? rowp[k0] : 0.0f;
                t[8 + i] = (k1 < K) ? rowp[k1] : 0.0f;
            }
#pragma unroll
            for (int j = 0; j < 8; ++j) fr.u32[j] = pack2bf(t[2 * j], t[2 * j + 1]);
        }
        lds[(size_t)p * 2]     = fr.u128[0];
        lds[(size_t)p * 2 + 1] = fr.u128[1];
    }
}

__device__ __forceinline__ BfPack lds_frag(const uint4* __restrict__ base, int idx) {
    BfPack a;
    a.u128[0] = base[(size_t)idx * 2];
    a.u128[1] = base[(size_t)idx * 2 + 1];
    return a;
}

// ---------------------------------------------------------------------------
// Main fused kernel: one block = 32 rows (two 16-row tiles), 8 waves;
// wave w owns cols 16w..16w+15 and both row tiles (B fragment reused 2x).
//
// Dynamic LDS layout (bytes):
//   [0      , 65536)  aLM[2][32kt][32lane][32B]   -- dead after LM GEMM, then:
//     [0    , 16384)  sL [32][128] f32
//     [16384, 32768)  sGp[32][128] f32
//     [32768, 49152)  sGn[32][128] f32
//   [65536 , 79872)   aGP[2][7][32][32B]
//   [72704 , 87040)   aGN[2][7][32][32B]
//   [79872 , 80000)   sRow[32] f32
// ---------------------------------------------------------------------------
#define SM_AGP   65536
#define SM_AGN   72704
#define SM_SROW  79872
#define SM_TOTAL 80000

__global__ void __launch_bounds__(256)
align_mi_main(const float* __restrict__ lm,  const float* __restrict__ gp,
              const float* __restrict__ gn,  const float* __restrict__ lm_b,
              const float* __restrict__ gnn_b,
              const unsigned* __restrict__ packLM,
              const unsigned* __restrict__ packGNN,
              float* __restrict__ partials) {
    extern __shared__ __align__(16) unsigned char smem[];
    uint4* aLM0 = reinterpret_cast<uint4*>(smem);                    // 32 KB
    uint4* aLM1 = aLM0 + 32 * 32 * 2;                                // 32 KB
    uint4* aGP0 = reinterpret_cast<uint4*>(smem + SM_AGP);           // 7 KB
    uint4* aGP1 = aGP0 + 7 * 32 * 2;                                 // 7 KB
    uint4* aGN0 = reinterpret_cast<uint4*>(smem + SM_AGN);           // 7 KB
    uint4* aGN1 = aGN0 + 7 * 32 * 2;                                 // 7 KB
    float (*sL)[128]  = reinterpret_cast<float (*)[128]>(smem);          // aliases aLM
    float (*sGp)[128] = reinterpret_cast<float (*)[128]>(smem + 16384);
    float (*sGn)[128] = reinterpret_cast<float (*)[128]>(smem + 32768);
    float* sRow = reinterpret_cast<float*>(smem + SM_SROW);

    const int w    = threadIdx.x >> 5;
    const int lane = threadIdx.x & 31;
    const int half = lane >> 4;
    const int l16  = lane & 15;
    const int rowbase = blockIdx.x * 32;
    const int colg = w * 16 + l16;

    // ---- Phase 0: stage bf16 A fragments for both row tiles ----
    stage_A(lm, 1024, 1024, 32, rowbase,      aLM0);
    stage_A(lm, 1024, 1024, 32, rowbase + 16, aLM1);
    stage_A(gp, 200,  200,  7,  rowbase,      aGP0);
    stage_A(gp, 200,  200,  7,  rowbase + 16, aGP1);
    stage_A(gn, 200,  200,  7,  rowbase,      aGN0);
    stage_A(gn, 200,  200,  7,  rowbase + 16, aGN1);
    __syncthreads();

    // ---- Phase 1: L tiles, K = 1024 -> 32 kt, 2 WMMAs per B fragment ----
    v8f c0 = {}, c1 = {};
    {
        const uint4* bbase = reinterpret_cast<const uint4*>(packLM);
#pragma unroll 4
        for (int kt = 0; kt < 32; ++kt) {
            BfPack bp;
            const uint4* bsrc = bbase + ((size_t)(kt * 8 + w) * 32 + lane) * 2;
            bp.u128[0] = bsrc[0];
            bp.u128[1] = bsrc[1];
            const BfPack a0 = lds_frag(aLM0, kt * 32 + lane);
            const BfPack a1 = lds_frag(aLM1, kt * 32 + lane);
            c0 = __builtin_amdgcn_wmma_f32_16x16x32_bf16(false, a0.v, false, bp.v,
                                                         (short)0, c0, false, false);
            c1 = __builtin_amdgcn_wmma_f32_16x16x32_bf16(false, a1.v, false, bp.v,
                                                         (short)0, c1, false, false);
        }
    }
    __syncthreads();   // all waves done reading aLM; results may overwrite it

    {
        const float bias = lm_b[colg];
#pragma unroll
        for (int r = 0; r < 8; ++r) {
            sL[r + half * 8][colg]      = c0[r] + bias;   // C/D: VGPR r -> row r (+8)
            sL[16 + r + half * 8][colg] = c1[r] + bias;
        }
    }

    // ---- Phase 2: Gp / Gn tiles, K = 200 padded to 224 -> 7 kt ----
    {
        const uint4* bbase = reinterpret_cast<const uint4*>(packGNN);
        const float bias = gnn_b[colg];

        v8f p0 = {}, p1 = {};
#pragma unroll
        for (int kt = 0; kt < 7; ++kt) {
            BfPack bp;
            const uint4* bsrc = bbase + ((size_t)(kt * 8 + w) * 32 + lane) * 2;
            bp.u128[0] = bsrc[0]; bp.u128[1] = bsrc[1];
            const BfPack a0 = lds_frag(aGP0, kt * 32 + lane);
            const BfPack a1 = lds_frag(aGP1, kt * 32 + lane);
            p0 = __builtin_amdgcn_wmma_f32_16x16x32_bf16(false, a0.v, false, bp.v,
                                                         (short)0, p0, false, false);
            p1 = __builtin_amdgcn_wmma_f32_16x16x32_bf16(false, a1.v, false, bp.v,
                                                         (short)0, p1, false, false);
        }
#pragma unroll
        for (int r = 0; r < 8; ++r) {
            sGp[r + half * 8][colg]      = p0[r] + bias;
            sGp[16 + r + half * 8][colg] = p1[r] + bias;
        }

        v8f n0 = {}, n1 = {};
#pragma unroll
        for (int kt = 0; kt < 7; ++kt) {
            BfPack bp;
            const uint4* bsrc = bbase + ((size_t)(kt * 8 + w) * 32 + lane) * 2;
            bp.u128[0] = bsrc[0]; bp.u128[1] = bsrc[1];
            const BfPack a0 = lds_frag(aGN0, kt * 32 + lane);
            const BfPack a1 = lds_frag(aGN1, kt * 32 + lane);
            n0 = __builtin_amdgcn_wmma_f32_16x16x32_bf16(false, a0.v, false, bp.v,
                                                         (short)0, n0, false, false);
            n1 = __builtin_amdgcn_wmma_f32_16x16x32_bf16(false, a1.v, false, bp.v,
                                                         (short)0, n1, false, false);
        }
#pragma unroll
        for (int r = 0; r < 8; ++r) {
            sGn[r + half * 8][colg]      = n0[r] + bias;
            sGn[16 + r + half * 8][colg] = n1[r] + bias;
        }
    }

    __syncthreads();

    // ---- Phase 3: per-row reductions (16 threads/row, 16 rows per pass) ----
#pragma unroll
    for (int rr = 0; rr < 2; ++rr) {
        const int row = rr * 16 + (threadIdx.x >> 4);
        const int seg = threadIdx.x & 15;
        float ll = 0.f, pp = 0.f, nn = 0.f, lp = 0.f, ln2 = 0.f;
#pragma unroll
        for (int j = 0; j < 8; ++j) {
            const int col = seg * 8 + j;
            const float l = sL[row][col], p = sGp[row][col], n = sGn[row][col];
            ll += l * l; pp += p * p; nn += n * n; lp += l * p; ln2 += l * n;
        }
#pragma unroll
        for (int off = 8; off >= 1; off >>= 1) {
            ll  += __shfl_xor(ll,  off, 16);
            pp  += __shfl_xor(pp,  off, 16);
            nn  += __shfl_xor(nn,  off, 16);
            lp  += __shfl_xor(lp,  off, 16);
            ln2 += __shfl_xor(ln2, off, 16);
        }
        if (seg == 0) {
            const float C = 1.0f / 11008.0f;
            const float cosp = lp  * rsqrtf(ll * pp);
            const float cosn = ln2 * rsqrtf(ll * nn);
            const float ep = expf(cosp);
            const float en = expf(cosn);
            const float pos = cosp - logf(ep + C);      // log(ep/(ep+C))
            const float neg = logf(C) - logf(en + C);   // log(1 - en/(en+C))
            sRow[row] = pos + neg;
        }
    }
    __syncthreads();
    if (threadIdx.x == 0) {
        float s = 0.f;
#pragma unroll
        for (int r = 0; r < 32; ++r) s += sRow[r];
        partials[blockIdx.x] = s;
    }
}

// Deterministic fixed-order final reduction over block partials.
__global__ void __launch_bounds__(256)
reduce_kernel(const float* __restrict__ partials, int n,
              float* __restrict__ out, float scale) {
    __shared__ float buf[256];
    float s = 0.f;
    for (int i = threadIdx.x; i < n; i += 256) s += partials[i];
    buf[threadIdx.x] = s;
    __syncthreads();
    for (int off = 128; off >= 1; off >>= 1) {
        if ((int)threadIdx.x < off) buf[threadIdx.x] += buf[threadIdx.x + off];
        __syncthreads();
    }
    if (threadIdx.x == 0) out[0] = scale * buf[0];
}

extern "C" void kernel_launch(void* const* d_in, const int* in_sizes, int n_in,
                              void* d_out, int out_size, void* d_ws, size_t ws_size,
                              hipStream_t stream) {
    const float* lm  = (const float*)d_in[0];   // [N,1024]
    const float* gp  = (const float*)d_in[1];   // [N,200]
    const float* gn  = (const float*)d_in[2];   // [N,200]
    const float* lmW = (const float*)d_in[3];   // [1024,128]
    const float* lmb = (const float*)d_in[4];   // [128]
    const float* gW  = (const float*)d_in[5];   // [200,128]
    const float* gb  = (const float*)d_in[6];   // [128]

    const int N = in_sizes[0] / 1024;
    const int nblocks = N / 32;

    unsigned char* ws = (unsigned char*)d_ws;
    const size_t lmPackBytes  = (size_t)32 * 8 * 1024;   // 256 KB
    const size_t gnnPackBytes = (size_t)7 * 8 * 1024;    //  56 KB
    unsigned* packLM  = (unsigned*)(ws);
    unsigned* packGNN = (unsigned*)(ws + lmPackBytes);
    float*    partials = (float*)(ws + lmPackBytes + gnnPackBytes);

    pack_weights_kernel<<<32, 256, 0, stream>>>(lmW, 1024, packLM);
    pack_weights_kernel<<<7,  256, 0, stream>>>(gW,  200,  packGNN);
    align_mi_main<<<nblocks, 256, SM_TOTAL, stream>>>(lm, gp, gn, lmb, gb,
                                                      packLM, packGNN, partials);
    reduce_kernel<<<1, 256, 0, stream>>>(partials, nblocks,
                                         (float*)d_out, -1.0f / (float)N);
}
